// Attention_28020366639391
// MI455X (gfx1250) — matrix operand order repaired and checked by reading, not compiled, not run on hardware
//
#include <hip/hip_runtime.h>
#include <hip/hip_bf16.h>
#include <math.h>

// ---------------- problem constants ----------------
#define B_ 8
#define S_ 1024
#define DM 768
#define NH 12
#define DHD 64

typedef __attribute__((ext_vector_type(16))) __bf16 v16bf;
typedef __attribute__((ext_vector_type(8)))  __bf16 v8bf;
typedef __attribute__((ext_vector_type(8)))  float  v8f;
typedef int v4i_ __attribute__((vector_size(16)));

__device__ __forceinline__ v16bf cat8(v8bf lo, v8bf hi) {
  return __builtin_shufflevector(lo, hi, 0,1,2,3,4,5,6,7,8,9,10,11,12,13,14,15);
}

__device__ __forceinline__ v8f wmma_bf16(v16bf a, v16bf b, v8f c) {
  // emits v_wmma_f32_16x16x32_bf16
  return __builtin_amdgcn_wmma_f32_16x16x32_bf16(false, a, false, b, (short)0, c, false, false);
}

__device__ __forceinline__ unsigned short bf16_bits(float f) {
  __bf16 h = (__bf16)f;
  return __builtin_bit_cast(unsigned short, h);
}

// ---- async global->LDS staging (gfx1250 GLOBAL_LOAD_ASYNC_TO_LDS_B128) ----
#if __has_builtin(__builtin_amdgcn_global_load_async_to_lds_b128)
#define HAVE_ASYNC 1
#else
#define HAVE_ASYNC 0
#endif

__device__ __forceinline__ void async_cp16(const void* g, void* l) {
#if HAVE_ASYNC
  __builtin_amdgcn_global_load_async_to_lds_b128(
      (__attribute__((address_space(1))) v4i_*)(void*)g,
      (__attribute__((address_space(3))) v4i_*)l,
      0, 0);
#else
  *(v8bf*)l = *(const v8bf*)g;
#endif
}

__device__ __forceinline__ void async_wait0() {
#if HAVE_ASYNC
  asm volatile("s_wait_asynccnt 0" ::: "memory");
#endif
}

// =====================================================================
// GEMM: Y[M,N] = (X[M,K] @ W[K,N] + bias[N]) * scale
// X,W,bias f32 in global; staged to LDS as bf16 (packed b32 stores);
// WMMA bf16 accumulate f32.
// Block = 256 thr (8 waves). Block tile 128Mx128N, K-step 64.
// Waves arranged 4(M)x2(N): each wave owns 32Mx64N = 2x4 C-fragments.
// =====================================================================
template<bool OUT_BF16>
__global__ __launch_bounds__(256)
void gemm_bias_kernel(const float* __restrict__ X,
                      const float* __restrict__ W,
                      const float* __restrict__ bias,
                      void* __restrict__ Y,
                      int M, int N, int K, float scale)
{
  __shared__ __align__(16) __bf16 sX[128 * 64];   // [m][k]
  __shared__ __align__(16) __bf16 sWT[128 * 64];  // [n][k]  (W transposed)

  const int tid      = threadIdx.x;
  const int lane     = tid & 31;
  const int wave     = tid >> 5;
  const int lane15   = lane & 15;
  const int laneHalf = lane >> 4;          // 0: lanes 0-15, 1: lanes 16-31

  const int m_blk = blockIdx.x * 128;
  const int n_blk = blockIdx.y * 128;
  const int wm = (wave >> 1) * 32;         // wave M offset: 0/32/64/96
  const int wn = (wave & 1) * 64;          // wave N offset: 0/64

  v8f acc[2][4];
  #pragma unroll
  for (int i = 0; i < 2; ++i)
    #pragma unroll
    for (int j = 0; j < 4; ++j) acc[i][j] = v8f{};

  for (int k0 = 0; k0 < K; k0 += 64) {
    __syncthreads();
    // stage X tile 128x64 (f32 -> bf16, paired b32 LDS stores)
    #pragma unroll
    for (int j = 0; j < 16; ++j) {
      int id = tid + 256 * j;              // 0..4095 pairs
      int r = id >> 5, c2 = (id & 31) << 1;
      const float* xp = &X[(size_t)(m_blk + r) * K + (k0 + c2)];
      unsigned int pk = (unsigned int)bf16_bits(xp[0]) |
                        ((unsigned int)bf16_bits(xp[1]) << 16);
      *(unsigned int*)&sX[r * 64 + c2] = pk;
    }
    // stage W tile 64x128 transposed into [n][k], paired along k
    #pragma unroll
    for (int j = 0; j < 16; ++j) {
      int id = tid + 256 * j;              // 0..4095 pairs
      int nn = id & 127, k2 = (id >> 7) << 1;
      float w0 = W[(size_t)(k0 + k2) * N + (n_blk + nn)];
      float w1 = W[(size_t)(k0 + k2 + 1) * N + (n_blk + nn)];
      unsigned int pk = (unsigned int)bf16_bits(w0) |
                        ((unsigned int)bf16_bits(w1) << 16);
      *(unsigned int*)&sWT[nn * 64 + k2] = pk;
    }
    if (k0 + 64 < K) {                     // global_prefetch_b8 for next tiles
      __builtin_prefetch(&X[(size_t)(m_blk + (tid >> 1)) * K + k0 + 64 + ((tid & 1) << 5)], 0, 0);
      __builtin_prefetch(&W[(size_t)(k0 + 64 + (tid >> 3)) * N + n_blk + ((tid & 7) << 4)], 0, 0);
    }
    __syncthreads();

    // A fragments (16x32 bf16, contiguous k = 16*laneHalf + s), 2 M-subtiles x 2 K-chunks
    v16bf afr[2][2];
    #pragma unroll
    for (int im = 0; im < 2; ++im) {
      const __bf16* row = &sX[(wm + im * 16 + lane15) * 64];
      #pragma unroll
      for (int c = 0; c < 2; ++c) {
        v8bf lo = *(const v8bf*)(row + c * 32 + laneHalf * 16);
        v8bf hi = *(const v8bf*)(row + c * 32 + laneHalf * 16 + 8);
        afr[im][c] = cat8(lo, hi);
      }
    }
    // B fragments (32x16 bf16): lanes 0-15: K=0..15 of col N=lane; 16-31: K=16..31.
    #pragma unroll
    for (int jn = 0; jn < 4; ++jn) {
      const __bf16* row = &sWT[(wn + jn * 16 + lane15) * 64];
      #pragma unroll
      for (int c = 0; c < 2; ++c) {
        v16bf bfr = *(const v16bf*)(row + c * 32 + laneHalf * 16);
        #pragma unroll
        for (int im = 0; im < 2; ++im)
          acc[im][jn] = wmma_bf16(afr[im][c], bfr, acc[im][jn]);
      }
    }
  }

  // epilogue: C layout — VGPR r = row (r + laneHalf*8), col = lane15
  #pragma unroll
  for (int jn = 0; jn < 4; ++jn) {
    int ncol = n_blk + wn + jn * 16 + lane15;
    float bv = bias[ncol];
    #pragma unroll
    for (int im = 0; im < 2; ++im) {
      #pragma unroll
      for (int r = 0; r < 8; ++r) {
        int mrow = m_blk + wm + im * 16 + r + laneHalf * 8;
        float val = (acc[im][jn][r] + bv) * scale;
        if (OUT_BF16) ((__bf16*)Y)[(size_t)mrow * N + ncol] = (__bf16)val;
        else          ((float*)Y)[(size_t)mrow * N + ncol] = val;
      }
    }
  }
}

// =====================================================================
// Flash attention for q < S/2 (unmasked rows). Q pre-scaled by 1/8.
// Block = 128 thr (4 waves), each wave owns one 16-row q tile.
// Grid: (B*H, (S/2)/64). K-step = 64 k-positions.
// K tile staged via GLOBAL_LOAD_ASYNC_TO_LDS_B128 (straight copy);
// V tile transposed manually with packed b32 stores.
// =====================================================================
__global__ __launch_bounds__(128)
void flash_kernel(const unsigned short* __restrict__ Qb_,
                  const unsigned short* __restrict__ Kb_,
                  const unsigned short* __restrict__ Vb_,
                  float* __restrict__ Ctx)
{
  __shared__ __align__(16) __bf16 sK[64 * 64];       // [kpos][dh]  (B [N][K] for QK^T)
  __shared__ __align__(16) __bf16 sVT[64 * 64];      // [dh][kpos]  (B [N][K] for PV)
  __shared__ __align__(16) __bf16 sP[4][16 * 64];    // per-wave P tile [q][kpos]

  const __bf16* Qb = (const __bf16*)Qb_;
  const __bf16* Kb = (const __bf16*)Kb_;

  const int tid      = threadIdx.x;
  const int lane     = tid & 31;
  const int wave     = tid >> 5;
  const int lane15   = lane & 15;
  const int laneHalf = lane >> 4;

  const int bh = blockIdx.x;            // 0..B*H-1
  const int b  = bh / NH;
  const int h  = bh % NH;
  const int q0 = blockIdx.y * 64 + wave * 16;    // q tile base (< S/2)

  const __bf16* Qbase = Qb + (size_t)b * S_ * DM + (size_t)h * DHD;
  const __bf16* Kbase = Kb + (size_t)b * S_ * DM + (size_t)h * DHD;
  const unsigned short* Vus = Vb_ + (size_t)b * S_ * DM + (size_t)h * DHD;

  // persistent Q A-fragments, dh chunks 0..31 and 32..63
  v16bf aQ[2];
  {
    const __bf16* qrow = Qbase + (size_t)(q0 + lane15) * DM;
    #pragma unroll
    for (int c = 0; c < 2; ++c) {
      v8bf lo = *(const v8bf*)(qrow + c * 32 + laneHalf * 16);
      v8bf hi = *(const v8bf*)(qrow + c * 32 + laneHalf * 16 + 8);
      aQ[c] = cat8(lo, hi);
    }
  }

  v8f accv[4];
  #pragma unroll
  for (int i = 0; i < 4; ++i) accv[i] = v8f{};
  float mrow[8], lrow[8];
  #pragma unroll
  for (int r = 0; r < 8; ++r) { mrow[r] = -INFINITY; lrow[r] = 0.f; }

  for (int kt = 0; kt < S_; kt += 64) {
    __syncthreads();
    // --- K tile 64x64 bf16: async copy, 512 x 16B chunks, 4 per thread ---
    #pragma unroll
    for (int j = 0; j < 4; ++j) {
      int id = tid + 128 * j;           // 0..511
      int r = id >> 3, c8 = (id & 7) << 3;
      async_cp16(Kbase + (size_t)(kt + r) * DM + c8, &sK[r * 64 + c8]);
    }
    // --- V tile 64x64 transposed, paired rows -> b32 LDS stores ---
    #pragma unroll
    for (int j = 0; j < 16; ++j) {
      int id = tid + 128 * j;           // 0..2047 pairs
      int c = id & 63, r2 = (id >> 6) << 1;
      unsigned int v0 = Vus[(size_t)(kt + r2) * DM + c];
      unsigned int v1 = Vus[(size_t)(kt + r2 + 1) * DM + c];
      *(unsigned int*)&sVT[c * 64 + r2] = v0 | (v1 << 16);
    }
    async_wait0();
    __syncthreads();

    // scores: 4 N-blocks (kpos 16-groups), each chained over dh (2 WMMAs)
    v8f sfr[4];
    #pragma unroll
    for (int nb = 0; nb < 4; ++nb) {
      v8f s = v8f{};
      #pragma unroll
      for (int c = 0; c < 2; ++c) {
        const __bf16* krow = &sK[(nb * 16 + lane15) * 64 + c * 32 + laneHalf * 16];
        v16bf bK = *(const v16bf*)krow;
        s = wmma_bf16(aQ[c], bK, s);
      }
      sfr[nb] = s;
    }

    // online softmax (lanes 0-15 <-> rows 0-7, 16-31 <-> rows 8-15;
    // xor masks 1/2/4/8 reduce within each 16-lane half)
    float alpha[8];
    #pragma unroll
    for (int r = 0; r < 8; ++r) {
      float mt = fmaxf(fmaxf(sfr[0][r], sfr[1][r]), fmaxf(sfr[2][r], sfr[3][r]));
      mt = fmaxf(mt, __shfl_xor(mt, 1, 32));
      mt = fmaxf(mt, __shfl_xor(mt, 2, 32));
      mt = fmaxf(mt, __shfl_xor(mt, 4, 32));
      mt = fmaxf(mt, __shfl_xor(mt, 8, 32));
      float mnew = fmaxf(mrow[r], mt);
      float p0 = exp2f((sfr[0][r] - mnew) * 1.44269504f);
      float p1 = exp2f((sfr[1][r] - mnew) * 1.44269504f);
      float p2 = exp2f((sfr[2][r] - mnew) * 1.44269504f);
      float p3 = exp2f((sfr[3][r] - mnew) * 1.44269504f);
      float ls = (p0 + p1) + (p2 + p3);
      ls += __shfl_xor(ls, 1, 32);
      ls += __shfl_xor(ls, 2, 32);
      ls += __shfl_xor(ls, 4, 32);
      ls += __shfl_xor(ls, 8, 32);
      float a = exp2f((mrow[r] - mnew) * 1.44269504f);
      lrow[r]  = lrow[r] * a + ls;
      mrow[r]  = mnew;
      alpha[r] = a;
      sfr[0][r] = p0; sfr[1][r] = p1; sfr[2][r] = p2; sfr[3][r] = p3;
    }

    // P: C layout -> LDS [q][kpos] (per-wave buffer; same-wave ordering only)
    __bf16* pbuf = &sP[wave][0];
    #pragma unroll
    for (int r = 0; r < 8; ++r) {
      int qr = r + laneHalf * 8;
      #pragma unroll
      for (int f = 0; f < 4; ++f)
        pbuf[qr * 64 + f * 16 + lane15] = (__bf16)sfr[f][r];
    }
    asm volatile("s_wait_dscnt 0" ::: "memory");   // make P visible to own wave

    // rescale accumulators while P settles
    #pragma unroll
    for (int nb = 0; nb < 4; ++nb)
      #pragma unroll
      for (int r = 0; r < 8; ++r)
        accv[nb][r] *= alpha[r];

    // P A-fragments (16x64 -> 2 chunks of K=32)
    v16bf aP[2];
    {
      const __bf16* prow = pbuf + (size_t)lane15 * 64;
      #pragma unroll
      for (int c = 0; c < 2; ++c) {
        v8bf lo = *(const v8bf*)(prow + c * 32 + laneHalf * 16);
        v8bf hi = *(const v8bf*)(prow + c * 32 + laneHalf * 16 + 8);
        aP[c] = cat8(lo, hi);
      }
    }
    // acc += P(16x64) x V(64x64): 4 dh N-blocks x 2 kpos chunks
    #pragma unroll
    for (int nb = 0; nb < 4; ++nb) {
      const __bf16* vrow = &sVT[(nb * 16 + lane15) * 64];
      #pragma unroll
      for (int c = 0; c < 2; ++c) {
        v16bf bV = *(const v16bf*)(vrow + c * 32 + laneHalf * 16);
        accv[nb] = wmma_bf16(aP[c], bV, accv[nb]);
      }
    }
  }

  // normalize and write ctx (f32)
  float* Cbase = Ctx + (size_t)b * S_ * DM + (size_t)h * DHD;
  #pragma unroll
  for (int r = 0; r < 8; ++r) {
    float inv = 1.f / lrow[r];
    int qr = q0 + r + laneHalf * 8;
    #pragma unroll
    for (int nb = 0; nb < 4; ++nb)
      Cbase[(size_t)qr * DM + nb * 16 + lane15] = accv[nb][r] * inv;
  }
}

// =====================================================================
// Masked rows (q >= S/2): scores zeroed pre-softmax -> uniform softmax
// -> ctx = mean(V). One block per (b,h).
// =====================================================================
__global__ __launch_bounds__(256)
void meanv_kernel(const unsigned short* __restrict__ Vb_, float* __restrict__ Ctx)
{
  __shared__ float red[256];
  const __bf16* Vb = (const __bf16*)Vb_;
  const int bh = blockIdx.x;
  const int b = bh / NH, h = bh % NH;
  const int tid = threadIdx.x;
  const int d   = tid & 63;
  const int grp = tid >> 6;                 // 0..3
  const __bf16* Vbase = Vb + (size_t)b * S_ * DM + (size_t)h * DHD;

  float sum = 0.f;
  for (int r = grp; r < S_; r += 4)
    sum += (float)Vbase[(size_t)r * DM + d];
  red[tid] = sum;
  __syncthreads();
  if (grp == 0)
    red[d] = (red[d] + red[64 + d] + red[128 + d] + red[192 + d]) * (1.f / (float)S_);
  __syncthreads();
  float mval = red[d];

  float* Cbase = Ctx + (size_t)b * S_ * DM + (size_t)h * DHD;
  for (int r = S_ / 2 + grp; r < S_; r += 4)
    Cbase[(size_t)r * DM + d] = mval;
}

// =====================================================================
extern "C" void kernel_launch(void* const* d_in, const int* in_sizes, int n_in,
                              void* d_out, int out_size, void* d_ws, size_t ws_size,
                              hipStream_t stream)
{
  (void)in_sizes; (void)n_in; (void)out_size; (void)ws_size;

  const float* X  = (const float*)d_in[0];
  const float* Wq = (const float*)d_in[1];
  const float* bq = (const float*)d_in[2];
  const float* Wk = (const float*)d_in[3];
  const float* bk = (const float*)d_in[4];
  const float* Wv = (const float*)d_in[5];
  const float* bv = (const float*)d_in[6];
  const float* Wo = (const float*)d_in[7];
  const float* bo = (const float*)d_in[8];
  float* out = (float*)d_out;

  const int M = B_ * S_;                            // 8192
  const size_t elems = (size_t)M * DM;              // 6,291,456

  char* ws = (char*)d_ws;
  unsigned short* Qb = (unsigned short*)ws;                    // bf16, Q*0.125 folded
  unsigned short* Kb = Qb + elems;                             // bf16
  unsigned short* Vb = Kb + elems;                             // bf16
  float*          Ctx = (float*)(Vb + elems);                  // f32 [B,S,DM]

  dim3 ggrid(M / 128, DM / 128);                    // 64 x 6

  // QKV projections (scale 1/sqrt(DH)=0.125 folded into Q, applied after bias)
  gemm_bias_kernel<true><<<ggrid, 256, 0, stream>>>(X, Wq, bq, (void*)Qb, M, DM, DM, 0.125f);
  gemm_bias_kernel<true><<<ggrid, 256, 0, stream>>>(X, Wk, bk, (void*)Kb, M, DM, DM, 1.0f);
  gemm_bias_kernel<true><<<ggrid, 256, 0, stream>>>(X, Wv, bv, (void*)Vb, M, DM, DM, 1.0f);

  // attention: real softmax for q < 512, mean(V) broadcast for q >= 512
  flash_kernel<<<dim3(B_ * NH, (S_ / 2) / 64), 128, 0, stream>>>(Qb, Kb, Vb, Ctx);
  meanv_kernel<<<B_ * NH, 256, 0, stream>>>(Vb, Ctx);

  // output projection -> f32
  gemm_bias_kernel<false><<<ggrid, 256, 0, stream>>>(Ctx, Wo, bo, (void*)out, M, DM, DM, 1.0f);
}
